// GATLayer_8615704396552
// MI455X (gfx1250) — matrix operand order repaired and checked by reading, not compiled, not run on hardware
//
#include <hip/hip_runtime.h>
#include <stdint.h>

#define GAT_N    16384
#define GAT_IN   128
#define GAT_OUT  64
#define K2_WAVES 8
#define JSPAN    (GAT_N / K2_WAVES)   // 2048 columns per wave
#define CHUNK    32                   // WMMA K step (bf16 16x16x32)
#define LOG2E    1.4426950408889634f

typedef __attribute__((ext_vector_type(16))) __bf16  v16bf;
typedef __attribute__((ext_vector_type(16))) short   v16s;
typedef __attribute__((ext_vector_type(8)))  float   v8f;

union V16 { v16s s; v16bf b; uint32_t u[8]; };

// native base-2 exponential: v_exp_f32
__device__ __forceinline__ float fexp2(float x) {
  return __builtin_amdgcn_exp2f(x);
}

__device__ __forceinline__ short f2bf(float f) {
  // round-to-nearest-even f32 -> bf16
  uint32_t u = __float_as_uint(f);
  uint32_t r = u + 0x7FFFu + ((u >> 16) & 1u);
  return (short)(r >> 16);
}

// pack two f32 -> packed bf16 dword with one v_perm_b32
__device__ __forceinline__ uint32_t pack2bf(float lo, float hi) {
  uint32_t ul = __float_as_uint(lo); ul += 0x7FFFu + ((ul >> 16) & 1u);
  uint32_t uh = __float_as_uint(hi); uh += 0x7FFFu + ((uh >> 16) & 1u);
  // result bytes: [1:0] = ul[3:2], [3:2] = uh[3:2]
  return __builtin_amdgcn_perm(uh, ul, 0x07060302u);
}

// ---------------------------------------------------------------------------
// Kernel 1: Wh = x@W (f32), f_i/f_j pre-scaled by log2(e) (valid since
// leaky_relu commutes with positive scaling), plus Wh packed as bf16 in the
// WMMA B-matrix lane layout:
//   B (32x16, 16-bit): lanes 0-15 hold K=0..7,16..23, lanes 16-31 hold K=8..15,24..31,
//   element idx within lane = (K & 7) + 8*(K >> 4), column n = lane & 15.
//   whp index = ((chunk*4 + t)*32 + lane)*16 + idx   (chunk = j/32, t = col/16)
// ---------------------------------------------------------------------------
__global__ __launch_bounds__(64) void gat_prep(
    const float* __restrict__ x, const float* __restrict__ W,
    const float* __restrict__ aw,
    float* __restrict__ fi, float* __restrict__ fj,
    short* __restrict__ whp)
{
  const int i = blockIdx.x;     // row 0..N-1
  const int c = threadIdx.x;    // col 0..63
  __shared__ float xs[GAT_IN];
  __shared__ float r2[4];

  xs[c]      = x[(size_t)i * GAT_IN + c];
  xs[c + 64] = x[(size_t)i * GAT_IN + 64 + c];
  __syncthreads();

  float wh = 0.f;
  #pragma unroll 8
  for (int k = 0; k < GAT_IN; ++k)
    wh = fmaf(xs[k], W[k * GAT_OUT + c], wh);

  float pi = wh * aw[c];
  float pj = wh * aw[GAT_OUT + c];
  #pragma unroll
  for (int o = 16; o > 0; o >>= 1) {
    pi += __shfl_xor(pi, o, 32);
    pj += __shfl_xor(pj, o, 32);
  }
  if ((c & 31) == 0) { r2[(c >> 5) * 2] = pi; r2[(c >> 5) * 2 + 1] = pj; }
  __syncthreads();
  if (c == 0) fi[i] = (r2[0] + r2[2]) * LOG2E;   // log2-domain scores
  if (c == 1) fj[i] = (r2[1] + r2[3]) * LOG2E;

  const int K     = i & 31;
  const int chunk = i >> 5;
  const int lane  = (c & 15) + 16 * ((K >> 3) & 1);
  const int idx   = (K & 7) + 8 * (K >> 4);
  const int t     = c >> 4;
  whp[(((size_t)(chunk * 4 + t) * 32 + lane) * 16) + idx] = f2bf(wh);
}

// ---------------------------------------------------------------------------
// Kernel 2: fused masked-softmax attention, online (flash-style), base-2.
// Block = 16-row tile, 8 waves split the 16384-column range.
// A-matrix (16x32 bf16) per lane: row = lane&15, kb = 8*(lane>>4),
//   element idx -> K = (idx<8) ? kb+idx : kb+16+(idx-8).
// ---------------------------------------------------------------------------
__global__ __launch_bounds__(256) void gat_attn(
    const int* __restrict__ adj,
    const float* __restrict__ fi,
    const float* __restrict__ fj,
    const short* __restrict__ whp,
    float* __restrict__ out)
{
  const int tid    = threadIdx.x;
  const int wave   = tid >> 5;
  const int lane   = tid & 31;
  const int laneHi = lane >> 4;
  const int row16  = lane & 15;
  const int row0   = blockIdx.x * 16;
  const int kb     = laneHi * 8;

  __shared__ float s_m[K2_WAVES][16];
  __shared__ float s_l[K2_WAVES][16];
  __shared__ float s_acc[K2_WAVES][16][GAT_OUT];

  const float fi_r   = fi[row0 + row16];
  const int* adjRow  = adj + (size_t)(row0 + row16) * GAT_N;
  const float NEGINF = -__builtin_inff();

  v8f c0 = {}, c1 = {}, c2 = {}, c3 = {};
  float m = -1e30f;   // finite init so rescale factor is never NaN
  float l = 0.f;

  const int jbase = wave * JSPAN;
  for (int j0 = jbase; j0 < jbase + JSPAN; j0 += CHUNK) {
    __builtin_prefetch(adjRow + j0 + CHUNK + kb, 0, 1);  // global_prefetch_b8

    const int4*   ap  = (const int4*)(adjRow + j0 + kb);
    int4 a0 = ap[0], a1 = ap[1];
    const int4*   ap2 = (const int4*)(adjRow + j0 + kb + 16);
    int4 a2 = ap2[0], a3 = ap2[1];
    const float4* gp  = (const float4*)(fj + j0 + kb);
    float4 g0 = gp[0], g1 = gp[1];
    const float4* gp2 = (const float4*)(fj + j0 + kb + 16);
    float4 g2 = gp2[0], g3 = gp2[1];

    int   ad[16] = {a0.x,a0.y,a0.z,a0.w, a1.x,a1.y,a1.z,a1.w,
                    a2.x,a2.y,a2.z,a2.w, a3.x,a3.y,a3.z,a3.w};
    float gv[16] = {g0.x,g0.y,g0.z,g0.w, g1.x,g1.y,g1.z,g1.w,
                    g2.x,g2.y,g2.z,g2.w, g3.x,g3.y,g3.z,g3.w};

    float s[16];
    float cmax = NEGINF;
    #pragma unroll
    for (int k = 0; k < 16; ++k) {
      float arg = fi_r + gv[k];
      float lr  = fmaxf(arg, 0.02f * arg);   // leaky_relu (log2 units)
      s[k] = (ad[k] > 0) ? lr : NEGINF;      // mask
      cmax = fmaxf(cmax, s[k]);
    }
    cmax = fmaxf(cmax, __shfl_xor(cmax, 16, 32));   // join row halves

    // online max update: rare after warm-up -> skip rescale on hot path
    if (__any(cmax > m)) {
      const float newm  = fmaxf(m, cmax);
      const float scale = fexp2(m - newm);
      l *= scale;
      #pragma unroll
      for (int v = 0; v < 8; ++v) {          // row of acc VGPR v is v+8*laneHi
        float sv = __shfl(scale, v + 8 * laneHi, 32);
        c0[v] *= sv; c1[v] *= sv; c2[v] *= sv; c3[v] *= sv;
      }
      m = newm;
    }

    float rs = 0.f;
    V16 pa;
    #pragma unroll
    for (int k = 0; k < 8; ++k) {
      float p0 = fexp2(s[2 * k]     - m);   // masked -> exp2(-inf) = 0
      float p1 = fexp2(s[2 * k + 1] - m);
      rs += p0 + p1;
      pa.u[k] = pack2bf(p0, p1);
    }
    rs += __shfl_xor(rs, 16, 32);
    l += rs;

    // B tiles (pre-packed, fully coalesced: 32B per lane per tile)
    const V16* bb = ((const V16*)whp) + ((size_t)(j0 >> 5) * 4) * 32 + lane;
    V16 b0 = bb[0], b1 = bb[32], b2 = bb[64], b3 = bb[96];

    c0 = __builtin_amdgcn_wmma_f32_16x16x32_bf16(false, pa.b, false, b0.b, (short)0, c0, false, false);
    c1 = __builtin_amdgcn_wmma_f32_16x16x32_bf16(false, pa.b, false, b1.b, (short)0, c1, false, false);
    c2 = __builtin_amdgcn_wmma_f32_16x16x32_bf16(false, pa.b, false, b2.b, (short)0, c2, false, false);
    c3 = __builtin_amdgcn_wmma_f32_16x16x32_bf16(false, pa.b, false, b3.b, (short)0, c3, false, false);
  }

  // publish per-wave partials
  if (laneHi == 0) { s_m[wave][row16] = m; s_l[wave][row16] = l; }
  #pragma unroll
  for (int v = 0; v < 8; ++v) {
    int r = v + 8 * laneHi;                  // C layout: M = vgpr + 8*(lane>=16)
    s_acc[wave][r][ 0 + row16] = c0[v];
    s_acc[wave][r][16 + row16] = c1[v];
    s_acc[wave][r][32 + row16] = c2[v];
    s_acc[wave][r][48 + row16] = c3[v];
  }
  __syncthreads();

  // combine 8 wave-partials (base-2 stats), normalize, ELU, store
  const int r  = tid >> 4;
  const int cb = (tid & 15) * 4;
  float M = -1e30f;
  #pragma unroll
  for (int w = 0; w < K2_WAVES; ++w) M = fmaxf(M, s_m[w][r]);
  float L = 0.f;
  float wfac[K2_WAVES];
  #pragma unroll
  for (int w = 0; w < K2_WAVES; ++w) {
    wfac[w] = fexp2(s_m[w][r] - M);
    L += s_l[w][r] * wfac[w];
  }
  const float invL = 1.f / L;
  #pragma unroll
  for (int q = 0; q < 4; ++q) {
    float a = 0.f;
    #pragma unroll
    for (int w = 0; w < K2_WAVES; ++w) a += s_acc[w][r][cb + q] * wfac[w];
    a *= invL;
    out[(size_t)(row0 + r) * GAT_OUT + cb + q] = (a > 0.f) ? a : (__expf(a) - 1.f);
  }
}

// ---------------------------------------------------------------------------
extern "C" void kernel_launch(void* const* d_in, const int* in_sizes, int n_in,
                              void* d_out, int out_size, void* d_ws, size_t ws_size,
                              hipStream_t stream) {
  (void)in_sizes; (void)n_in; (void)out_size; (void)ws_size;
  const float* x   = (const float*)d_in[0];
  const int*   adj = (const int*)d_in[1];
  const float* W   = (const float*)d_in[2];
  const float* aw  = (const float*)d_in[3];
  float* out = (float*)d_out;

  short* whp = (short*)d_ws;                                    // 2 MiB bf16-packed Wh
  float* fi  = (float*)((char*)d_ws + (size_t)GAT_N * GAT_OUT * 2);
  float* fj  = fi + GAT_N;

  gat_prep<<<GAT_N, 64, 0, stream>>>(x, W, aw, fi, fj, whp);
  gat_attn<<<GAT_N / 16, 256, 0, stream>>>(adj, fi, fj, whp, out);
}
